// MNISTFullyCKAN_57286273794762
// MI455X (gfx1250) — matrix operand order repaired and checked by reading, not compiled, not run on hardware
//
#include <hip/hip_runtime.h>
#include <hip/hip_bf16.h>
#include <math.h>

typedef _Float16 v16h  __attribute__((ext_vector_type(16)));
typedef _Float16 half8 __attribute__((ext_vector_type(8)));
typedef float    v8f   __attribute__((ext_vector_type(8)));

#define NB 8  // spline bases per element (GRID_SIZE=5, DEGREE=3)

// Cox-de Boor on the uniform efficient-kan grid: grid[i] = (i-3)*0.4 - 1, i=0..11
__device__ __forceinline__ void bspline8(float x, float* o) {
  const float h  = 0.4f;
  const float k0 = -2.2f;  // grid[0]
  float b[11];
#pragma unroll
  for (int i = 0; i < 11; ++i) {
    float g0 = k0 + i * h;
    b[i] = (x >= g0 && x < g0 + h) ? 1.0f : 0.0f;
  }
#pragma unroll
  for (int d = 1; d <= 3; ++d) {
    float inv = 1.0f / (d * h);
#pragma unroll
    for (int i = 0; i < 11 - d; ++i) {
      float gi = k0 + i * h;
      b[i] = (x - gi) * inv * b[i] + (gi + (d + 1) * h - x) * inv * b[i + 1];
    }
  }
#pragma unroll
  for (int i = 0; i < 8; ++i) o[i] = b[i];
}

// Fused KANConv layer: im2col gather + silu/bspline activation (f16 into LDS)
// + single fused WMMA GEMM over K = 8F (spline) + F (base), f32 accumulate.
// Fused K layout: kk in [0, 8F) -> spline(f=kk/8, j=kk%8); kk in [8F, 9F) -> silu(f=kk-8F).
template <int C, int H, int W, int KS, int PAD, int O, int WAVES>
__global__ void kan_conv_kernel(const float* __restrict__ in,
                                const float* __restrict__ base_w,
                                const float* __restrict__ spline_w,
                                float* __restrict__ out, int Bn) {
  constexpr int F    = C * KS * KS;
  constexpr int Ho   = H + 2 * PAD - KS + 1;
  constexpr int Wo   = W + 2 * PAD - KS + 1;
  constexpr int HoWo = Ho * Wo;                      // >= 16 for every layer
  constexpr int KROW = F * (1 + NB);
  constexpr int KPAD = ((KROW + 31) / 32) * 32;
  constexpr int OT   = (O + 15) / 16;
  constexpr int OT16 = OT * 16;
  constexpr int NTH  = WAVES * 32;

  extern __shared__ __align__(16) char smem_raw[];
  _Float16* smem = (_Float16*)smem_raw;
  _Float16* Bt   = smem;                                   // [OT16][KPAD], shared by block
  const int tid  = threadIdx.x;
  const int wave = tid >> 5;
  const int lane = tid & 31;
  _Float16* A    = smem + OT16 * KPAD + wave * 16 * KPAD;  // per-wave [16][KPAD]

  // Zero all LDS once (establishes K-padding / O-padding zeros permanently).
  {
    const int tot = (OT16 + WAVES * 16) * KPAD / 2;
    unsigned* p = (unsigned*)smem;
    for (int i = tid; i < tot; i += NTH) p[i] = 0u;
  }
  __syncthreads();

  // Pack fused weights Wt[n][kk] = [spline_w | base_w] as f16, transposed rows.
  for (int idx = tid; idx < O * KROW; idx += NTH) {
    const int n = idx / KROW, kk = idx - n * KROW;
    float v;
    if (kk < NB * F) {
      v = spline_w[(n * F + (kk >> 3)) * NB + (kk & 7)];
    } else {
      v = base_w[n * F + (kk - NB * F)];
    }
    Bt[n * KPAD + kk] = (_Float16)v;
  }
  __syncthreads();

  const int N      = Bn * HoWo;          // rows (all layer sizes divisible by 16)
  const int ntiles = N >> 4;
  const int stride = gridDim.x * WAVES;
  const int iters  = (ntiles + stride - 1) / stride;  // uniform across block
  const int g  = lane >> 4;
  const int ln = lane & 15;

  for (int it = 0; it < iters; ++it) {
    const int tile   = blockIdx.x * WAVES + wave + it * stride;
    const bool valid = tile < ntiles;    // wave-uniform

    // ---- A-tile fill: one row per lane (both halves share a row, split features).
    if (valid) {
      const int r   = tile * 16 + ln;
      const int b   = r / HoWo;                 // div by compile-time constant, once
      const int rem = r - b * HoWo;
      const int oy  = rem / Wo, ox = rem - oy * Wo;
      _Float16* row = A + ln * KPAD;
      const float* inb = in + b * (C * H * W);
      for (int f = g; f < F; f += 2) {          // g=0 even f, g=1 odd f
        const int c  = f / (KS * KS);
        const int kk = f - c * (KS * KS);
        const int ky = kk / KS, kx = kk - ky * KS;
        const int iy = oy - PAD + ky, ix = ox - PAD + kx;
        float p = 0.0f;
        if (iy >= 0 && iy < H && ix >= 0 && ix < W)
          p = inb[(c * H + iy) * W + ix];
        float bs[8];
        bspline8(p, bs);
        half8 pk;
#pragma unroll
        for (int j = 0; j < 8; ++j) pk[j] = (_Float16)bs[j];
        *(half8*)(row + f * NB) = pk;                       // 16B-aligned b128 store
        row[NB * F + f] = (_Float16)(p / (1.0f + __expf(-p)));
      }
      // Speculative prefetch of the batch block the next tile will touch.
      const int tn = tile + stride;
      if (tn < ntiles)
        __builtin_prefetch(in + (size_t)((tn * 16) / HoWo) * (C * H * W), 0, 1);
    }
    __syncthreads();

    // ---- WMMA K-loop (EXEC all ones: executed unconditionally by all lanes)
    v8f acc[OT];
#pragma unroll
    for (int ot = 0; ot < OT; ++ot)
      acc[ot] = (v8f){0.f, 0.f, 0.f, 0.f, 0.f, 0.f, 0.f, 0.f};

    const _Float16* arow = A + ln * KPAD;
    for (int kt = 0; kt < KPAD / 32; ++kt) {
      union { v16h v; half8 h[2]; } af;
      // A 16x32 f16 layout: group g holds K = 8g+{0..7} and 16+8g+{0..7}
      af.h[0] = *(const half8*)(arow + kt * 32 + g * 8);
      af.h[1] = *(const half8*)(arow + kt * 32 + g * 8 + 16);
#pragma unroll
      for (int ot = 0; ot < OT; ++ot) {
        union { v16h v; half8 h[2]; } bf;
        // B 32x16 f16 layout: lane = column n, group g holds K = 8g+{0..7} and 16+8g+{0..7}
        const _Float16* brow = Bt + (ot * 16 + ln) * KPAD + kt * 32 + g * 8;
        bf.h[0] = *(const half8*)(brow);
        bf.h[1] = *(const half8*)(brow + 16);
        acc[ot] = __builtin_amdgcn_wmma_f32_16x16x32_f16(
            false, af.v, false, bf.v, (short)0, acc[ot], false, false);
      }
    }

    // ---- scatter C fragment: VGPR v -> row (8g+v), lane&15 -> out channel.
    // One divide for the lane-group's first row, then wrap-once increments
    // (legal since HoWo >= 16: a 16-row tile crosses at most one boundary/step).
    if (valid) {
      int r   = tile * 16 + 8 * g;
      int b_  = r / HoWo;
      int rem = r - b_ * HoWo;
      int oy_ = rem / Wo, ox_ = rem - oy_ * Wo;
#pragma unroll
      for (int v = 0; v < 8; ++v) {
        const int base = (b_ * O * Ho + oy_) * Wo + ox_;
#pragma unroll
        for (int ot = 0; ot < OT; ++ot) {
          const int n = ot * 16 + ln;
          if (n < O) out[base + n * HoWo] = acc[ot][v];
        }
        if (++ox_ == Wo) { ox_ = 0; if (++oy_ == Ho) { oy_ = 0; ++b_; } }
      }
    }
    __syncthreads();
  }
}

// 2x2 max pool, stride 2, VALID (7 -> 3 drops last row/col like the reference)
__global__ void maxpool2_kernel(const float* __restrict__ in, float* __restrict__ out,
                                int BC, int H, int W) {
  const int Ho = H / 2, Wo = W / 2;
  const int total = BC * Ho * Wo;
  for (int i = blockIdx.x * blockDim.x + threadIdx.x; i < total;
       i += gridDim.x * blockDim.x) {
    const int bc  = i / (Ho * Wo);
    const int rem = i - bc * (Ho * Wo);
    const int y = rem / Wo, x = rem - y * Wo;
    const float* p = in + (bc * H + 2 * y) * W + 2 * x;
    out[i] = fmaxf(fmaxf(p[0], p[1]), fmaxf(p[W], p[W + 1]));
  }
}

// global average pool (4x4) + FC 32->10
__global__ void head_kernel(const float* __restrict__ h4, const float* __restrict__ fc_w,
                            const float* __restrict__ fc_b, float* __restrict__ out, int Bn) {
  const int b = blockIdx.x * blockDim.x + threadIdx.x;
  if (b >= Bn) return;
  float avg[32];
#pragma unroll
  for (int o = 0; o < 32; ++o) {
    const float* p = h4 + (b * 32 + o) * 16;
    float s = 0.f;
#pragma unroll
    for (int i = 0; i < 16; ++i) s += p[i];
    avg[o] = s * (1.0f / 16.0f);
  }
#pragma unroll
  for (int c = 0; c < 10; ++c) {
    float s = fc_b[c];
#pragma unroll
    for (int o = 0; o < 32; ++o) s += avg[o] * fc_w[c * 32 + o];
    out[b * 10 + c] = s;
  }
}

static inline int imin_(int a, int b) { return a < b ? a : b; }
constexpr int kpad_of(int F) { return ((F * 9 + 31) / 32) * 32; }
constexpr int sh_of(int F, int O, int Wv) {
  return (((O + 15) / 16) * 16 + Wv * 16) * kpad_of(F) * 2;
}

extern "C" void kernel_launch(void* const* d_in, const int* in_sizes, int n_in,
                              void* d_out, int out_size, void* d_ws, size_t ws_size,
                              hipStream_t stream) {
  (void)n_in; (void)out_size; (void)ws_size;
  const float* x   = (const float*)d_in[0];
  const float* bw1 = (const float*)d_in[1];
  const float* sw1 = (const float*)d_in[2];
  const float* bw2 = (const float*)d_in[3];
  const float* sw2 = (const float*)d_in[4];
  const float* bw3 = (const float*)d_in[5];
  const float* sw3 = (const float*)d_in[6];
  const float* bw4 = (const float*)d_in[7];
  const float* sw4 = (const float*)d_in[8];
  const float* fcw = (const float*)d_in[9];
  const float* fcb = (const float*)d_in[10];
  float* out = (float*)d_out;

  const int B = in_sizes[0] / (28 * 28);  // C=1 input

  // Ping-pong workspace: bufA sized for the largest tensor (h1: B*4*28*28)
  float* bufA = (float*)d_ws;
  float* bufB = bufA + (size_t)B * 4 * 28 * 28;

  // Layer 1: [B,1,28,28] -> [B,4,28,28]
  {
    const int ntiles = B * 28 * 28 / 16;
    const int grid = imin_((ntiles + 3) / 4, 4096);
    kan_conv_kernel<1, 28, 28, 3, 1, 4, 4>
        <<<grid, 128, sh_of(9, 4, 4), stream>>>(x, bw1, sw1, bufA, B);
  }
  { // pool -> [B,4,14,14]
    const int tot = B * 4 * 14 * 14;
    maxpool2_kernel<<<(tot + 255) / 256, 256, 0, stream>>>(bufA, bufB, B * 4, 28, 28);
  }
  // Layer 2: -> [B,8,14,14]
  {
    const int ntiles = B * 14 * 14 / 16;
    const int grid = imin_((ntiles + 3) / 4, 4096);
    kan_conv_kernel<4, 14, 14, 3, 1, 8, 4>
        <<<grid, 128, sh_of(36, 8, 4), stream>>>(bufB, bw2, sw2, bufA, B);
  }
  { // pool -> [B,8,7,7]
    const int tot = B * 8 * 7 * 7;
    maxpool2_kernel<<<(tot + 255) / 256, 256, 0, stream>>>(bufA, bufB, B * 8, 14, 14);
  }
  // Layer 3: -> [B,16,7,7]
  {
    const int ntiles = B * 7 * 7 / 16;
    const int grid = imin_((ntiles + 3) / 4, 4096);
    kan_conv_kernel<8, 7, 7, 3, 1, 16, 4>
        <<<grid, 128, sh_of(72, 16, 4), stream>>>(bufB, bw3, sw3, bufA, B);
  }
  { // pool -> [B,16,3,3]
    const int tot = B * 16 * 3 * 3;
    maxpool2_kernel<<<(tot + 255) / 256, 256, 0, stream>>>(bufA, bufB, B * 16, 7, 7);
  }
  // Layer 4: ksize=2, pad=1 -> [B,32,4,4]
  {
    const int ntiles = B * 4 * 4 / 16;
    const int grid = imin_((ntiles + 3) / 4, 4096);
    kan_conv_kernel<16, 3, 3, 2, 1, 32, 4>
        <<<grid, 128, sh_of(64, 32, 4), stream>>>(bufB, bw4, sw4, bufA, B);
  }
  // avg pool + FC -> [B,10]
  head_kernel<<<(B + 127) / 128, 128, 0, stream>>>(bufA, fcw, fcb, out, B);
}
